// SelfAttention_36532991820095
// MI455X (gfx1250) — matrix operand order repaired and checked by reading, not compiled, not run on hardware
//
#include <hip/hip_runtime.h>

// ---------------- vector types ----------------
typedef __attribute__((ext_vector_type(16))) __bf16 bf16x16;
typedef __attribute__((ext_vector_type(8)))  __bf16 bf16x8;
typedef __attribute__((ext_vector_type(4)))  __bf16 bf16x4;
typedef __attribute__((ext_vector_type(2)))  __bf16 bf16x2;
typedef __attribute__((ext_vector_type(8)))  float  f32x8;
typedef __attribute__((ext_vector_type(4)))  int    v4i;

#define DEVINL __device__ __forceinline__
#define AS1 __attribute__((address_space(1)))
#define AS3 __attribute__((address_space(3)))

// -------- CDNA5 async / tensor-DMA feature detection (device pass only) -----
#if defined(__gfx1250__)
#if __has_builtin(__builtin_amdgcn_global_load_async_to_lds_b128)
#define HAVE_ASYNC_LDS 1
#endif
#if __has_builtin(__builtin_amdgcn_tensor_load_to_lds)
#define HAVE_TDM 1
#endif
#endif

#ifndef HAVE_ASYNC_LDS
#define HAVE_ASYNC_LDS 0
#endif
#ifndef HAVE_TDM
#define HAVE_TDM 0
#endif

#if __has_include(<hip/amd_detail/amd_gfx1250_TDM.h>)
#define CDNA5_TDM_6ARG 1  // amdgpu-toolchain / therock headers -> 6-arg builtin
#else
#define CDNA5_TDM_6ARG 0  // ROCm 7.2 clang-22 -> 5-arg builtin
#endif

#if HAVE_ASYNC_LDS
DEVINL void wait_async0() {
#if __has_builtin(__builtin_amdgcn_s_wait_asynccnt)
  __builtin_amdgcn_s_wait_asynccnt(0);
#else
  asm volatile("s_wait_asynccnt 0" ::: "memory");
#endif
}
DEVINL void async_load_b128(const void* g, void* l) {
  // GLOBAL_LOAD_ASYNC_TO_LDS_B128: per-lane 16B global -> LDS (ASYNCcnt)
  // Builtin prototype (from compiler diagnostic): (v4i AS1*, v4i AS3*, Ii, Ii)
  __builtin_amdgcn_global_load_async_to_lds_b128((AS1 v4i*)(v4i*)(void*)g,
                                                 (AS3 v4i*)(v4i*)l, 0, 0);
}
#endif

#if HAVE_TDM
typedef __attribute__((ext_vector_type(4))) unsigned int u32x4;
typedef __attribute__((ext_vector_type(8))) int i32x8;
typedef __attribute__((ext_vector_type(4))) int i32x4;

DEVINL void wait_tensor0() {
#if __has_builtin(__builtin_amdgcn_s_wait_tensorcnt)
  __builtin_amdgcn_s_wait_tensorcnt(0);
#else
  asm volatile("s_wait_tensorcnt 0" ::: "memory");
#endif
}
DEVINL unsigned lds_byte_addr(const void* p) {
  return (unsigned)(unsigned long long)(AS3 const void*)p;
}
// 2D bf16 tile load via the Tensor Data Mover.
// tile_d0 elements contiguous (one row), tile_d1 rows, row stride in elements.
// LDS padding: after every 16 DWORDs (64B row) insert 4 DWORDs (16B) -> 80B
// LDS row pitch, matching the [.][40] __shared__ tiles below.
DEVINL void tdm_load_2d(unsigned lds, const void* gaddr, unsigned tensor_d0,
                        unsigned stride0_elems, unsigned tile_d0,
                        unsigned tile_d1) {
  unsigned long long ga = (unsigned long long)gaddr;
  u32x4 g0;
  g0[0] = 1u;                                            // count=1 (valid D#)
  g0[1] = lds;                                           // lds_addr bytes
  g0[2] = (unsigned)ga;                                  // global_addr lo
  g0[3] = (unsigned)((ga >> 32) & 0x01ffffffu) | (2u << 30);  // hi + type=2
  const unsigned td1 = 1u << 20;                         // generous dim1
  i32x8 g1;
  g1[0] = (int)((1u << 16) | (1u << 20) | (3u << 22) | (3u << 25));
  //            data_size=2B  pad_enable   every 16 DW    pad 4 DW
  g1[1] = (int)((tensor_d0 & 0xffffu) << 16);
  g1[2] = (int)(((tensor_d0 >> 16) & 0xffffu) | ((td1 & 0xffffu) << 16));
  g1[3] = (int)(((td1 >> 16) & 0xffffu) | ((tile_d0 & 0xffffu) << 16));
  g1[4] = (int)(tile_d1 & 0xffffu);                      // tile_dim1, dim2=0
  g1[5] = (int)stride0_elems;                            // dim0 stride lo
  g1[6] = 0;                                             // stride hi, s1 lo
  g1[7] = 0;
  i32x4 z4 = {0, 0, 0, 0};
#if CDNA5_TDM_6ARG
  i32x8 z8 = {0, 0, 0, 0, 0, 0, 0, 0};
  __builtin_amdgcn_tensor_load_to_lds(g0, g1, z4, z4, z8, 0);
#else
  __builtin_amdgcn_tensor_load_to_lds(g0, g1, z4, z4, 0);
#endif
}
#endif

DEVINL f32x8 wmma_bf16(bf16x16 a, bf16x16 b, f32x8 c) {
  // D = A(16x32 bf16) * B(32x16 bf16) + C(16x16 f32)
  return __builtin_amdgcn_wmma_f32_16x16x32_bf16(false, a, false, b, (short)0, c,
                                                 false, false);
}

DEVINL bf16x16 join8(bf16x8 lo, bf16x8 hi) {
  return __builtin_shufflevector(lo, hi, 0, 1, 2, 3, 4, 5, 6, 7, 8, 9, 10, 11,
                                 12, 13, 14, 15);
}

// ---------------- fp32 -> bf16 conversion ----------------
__global__ __launch_bounds__(256) void cvt_f32_to_bf16_x4(
    const float4* __restrict__ s, bf16x4* __restrict__ d, int n4) {
  int i = blockIdx.x * 256 + threadIdx.x;
  if (i < n4) {
    float4 v = s[i];
    bf16x4 o = {(__bf16)v.x, (__bf16)v.y, (__bf16)v.z, (__bf16)v.w};
    d[i] = o;
  }
}

// ---------------- GEMM: C[4096,1024] = A[4096,1024] @ B[1024,1024]^T ---------
// ALAYOUT: 0 = plain row-major bf16, 1 = attn-out layout [b,h,t,64]
// EPI:     0 = scale + bf16 store into [b,h,t,64] layout, 1 = f32 + bias -> Cf
template <int ALAYOUT, int EPI>
__global__ __launch_bounds__(256) void gemm_bf16_k(
    const __bf16* __restrict__ A, const __bf16* __restrict__ B,
    __bf16* __restrict__ Cb, float* __restrict__ Cf,
    const float* __restrict__ bias, float scale) {
  // LDS tiles padded to 40 bf16 (80B) rows: conflict-free b128 fragment reads
  __shared__ __bf16 As[2][128][40];
  __shared__ __bf16 Bs[2][128][40];

  const int tid = threadIdx.x;
  const int lane = tid & 31, wave = tid >> 5;
  const int half = lane >> 4, l15 = lane & 15;
  const int waveM = wave >> 1, waveN = wave & 1;
  const int mBase = blockIdx.y * 128, nBase = blockIdx.x * 128;
  const int lr0 = tid >> 2, lkc = tid & 3;  // staging row / 8-elem chunk

  auto aaddr = [&](int m, int k8) -> const uint4* {
    if constexpr (ALAYOUT == 0) {
      return (const uint4*)(A + (size_t)m * 1024 + (size_t)k8 * 8);
    } else {
      // feature index k = h*64 + d ; 32-wide k-chunks never cross a head block
      int head = k8 >> 3, d0 = (k8 & 7) * 8;
      int b = m >> 11, tt = m & 2047;
      return (const uint4*)(A + ((((size_t)b * 16 + head) * 2048 + tt) * 64 + d0));
    }
  };

#if !HAVE_TDM && !HAVE_ASYNC_LDS
  uint4 ra0, ra1, rb0, rb1;
#endif

  // Issue the DMA/loads for k-tile kt into LDS buffer `buf`.
  auto stage_issue = [&](int kt, int buf) {
#if HAVE_TDM
    if (wave == 0) {
      const __bf16* ga;
      unsigned td0, st0;
      if constexpr (ALAYOUT == 0) {
        ga = A + (size_t)mBase * 1024 + (size_t)kt * 32;
        td0 = 1024; st0 = 1024;
      } else {
        int head = kt >> 1, d0 = (kt & 1) * 32;
        int b = mBase >> 11, tt = mBase & 2047;
        ga = A + (((size_t)b * 16 + head) * 2048 + tt) * 64 + d0;
        td0 = 64; st0 = 64;
      }
      tdm_load_2d(lds_byte_addr(&As[buf][0][0]), ga, td0, st0, 32, 128);
      tdm_load_2d(lds_byte_addr(&Bs[buf][0][0]),
                  B + (size_t)nBase * 1024 + (size_t)kt * 32, 1024, 1024, 32,
                  128);
    }
#elif HAVE_ASYNC_LDS
    int k8 = kt * 4 + lkc;
    async_load_b128(aaddr(mBase + lr0, k8), &As[buf][lr0][lkc * 8]);
    async_load_b128(aaddr(mBase + lr0 + 64, k8), &As[buf][lr0 + 64][lkc * 8]);
    async_load_b128(B + (size_t)(nBase + lr0) * 1024 + (size_t)k8 * 8,
                    &Bs[buf][lr0][lkc * 8]);
    async_load_b128(B + (size_t)(nBase + lr0 + 64) * 1024 + (size_t)k8 * 8,
                    &Bs[buf][lr0 + 64][lkc * 8]);
#else
    int k8 = kt * 4 + lkc;
    ra0 = *aaddr(mBase + lr0, k8);
    ra1 = *aaddr(mBase + lr0 + 64, k8);
    rb0 = *(const uint4*)(B + (size_t)(nBase + lr0) * 1024 + (size_t)k8 * 8);
    rb1 = *(const uint4*)(B + (size_t)(nBase + lr0 + 64) * 1024 + (size_t)k8 * 8);
    (void)buf;
#endif
  };
  // Complete the staging for buffer `buf` (called after the math of this iter).
  auto stage_wait = [&](int buf) {
#if HAVE_TDM
    if (wave == 0) wait_tensor0();
    (void)buf;
#elif HAVE_ASYNC_LDS
    wait_async0();
    (void)buf;
#else
    *(uint4*)&As[buf][lr0][lkc * 8] = ra0;
    *(uint4*)&As[buf][lr0 + 64][lkc * 8] = ra1;
    *(uint4*)&Bs[buf][lr0][lkc * 8] = rb0;
    *(uint4*)&Bs[buf][lr0 + 64][lkc * 8] = rb1;
#endif
  };

  f32x8 acc[2][4] = {};
  stage_issue(0, 0);
  stage_wait(0);
  __syncthreads();

  for (int kt = 0; kt < 32; ++kt) {
    const int cur = kt & 1;
    if (kt + 1 < 32) {
      stage_issue(kt + 1, cur ^ 1);
#if !HAVE_TDM
      if (kt + 2 < 32) {
        __builtin_prefetch(aaddr(mBase + lr0, (kt + 2) * 4 + lkc), 0, 1);
        __builtin_prefetch(
            B + (size_t)(nBase + lr0) * 1024 + (size_t)((kt + 2) * 4 + lkc) * 8,
            0, 1);
      }
#endif
    }
    bf16x16 afr[2], bfr[4];
#pragma unroll
    for (int mt = 0; mt < 2; ++mt) {
      const __bf16* p = &As[cur][waveM * 32 + mt * 16 + l15][half * 8];
      afr[mt] = join8(*(const bf16x8*)p, *(const bf16x8*)(p + 16));
    }
#pragma unroll
    for (int nt = 0; nt < 4; ++nt) {
      const __bf16* p = &Bs[cur][waveN * 64 + nt * 16 + l15][half * 8];
      bfr[nt] = join8(*(const bf16x8*)p, *(const bf16x8*)(p + 16));
    }
#pragma unroll
    for (int mt = 0; mt < 2; ++mt)
#pragma unroll
      for (int nt = 0; nt < 4; ++nt)
        acc[mt][nt] = wmma_bf16(afr[mt], bfr[nt], acc[mt][nt]);

    if (kt + 1 < 32) stage_wait(cur ^ 1);
    __syncthreads();
  }

  // epilogue: C layout -> lane holds column n = l15, rows m = base + half*8 + r
#pragma unroll
  for (int mt = 0; mt < 2; ++mt) {
#pragma unroll
    for (int nt = 0; nt < 4; ++nt) {
      const int n = nBase + waveN * 64 + nt * 16 + l15;
      if constexpr (EPI == 0) {
        const int head = n >> 6, d = n & 63;
#pragma unroll
        for (int r = 0; r < 8; ++r) {
          const int m = mBase + waveM * 32 + mt * 16 + half * 8 + r;
          const int b = m >> 11, tt = m & 2047;
          Cb[(((size_t)b * 16 + head) * 2048 + tt) * 64 + d] =
              (__bf16)(acc[mt][nt][r] * scale);
        }
      } else {
        const float bv = bias[n];
#pragma unroll
        for (int r = 0; r < 8; ++r) {
          const int m = mBase + waveM * 32 + mt * 16 + half * 8 + r;
          Cf[(size_t)m * 1024 + n] = acc[mt][nt][r] + bv;
        }
      }
    }
  }
}

// ---------------- flash attention over [b,h,t,64] bf16 workspaces ------------
// Computes S^T = K*Q^T (softmax over keys stays in-lane), then O^T = V^T * P^T.
__global__ __launch_bounds__(256) void attn_kernel(
    const __bf16* __restrict__ Q, const __bf16* __restrict__ Kk,
    const __bf16* __restrict__ V, __bf16* __restrict__ O) {
  __shared__ __bf16 Qs[128][72];    // [q][d]
  __shared__ __bf16 Ks[64][72];     // [kv][d]
  __shared__ __bf16 Vt[64][72];     // [d][kv]  (transposed at staging)
  __shared__ __bf16 Ps[8][16][72];  // per-wave P: [q][kv]

  const int tid = threadIdx.x;
  const int lane = tid & 31, wave = tid >> 5;
  const int half = lane >> 4, l15 = lane & 15;
  const int bh = blockIdx.x;        // 0..31 (batch*head)
  const int q0 = blockIdx.y * 128;  // q block base
  const size_t base = (size_t)bh * 2048 * 64;

  // stage Q block (128 x 64): 1024 8-elem chunks, 4 per thread
#pragma unroll
  for (int i = 0; i < 4; ++i) {
    int c = tid + i * 256;
    int row = c >> 3, ch = c & 7;
#if HAVE_ASYNC_LDS
    async_load_b128(Q + base + (size_t)(q0 + row) * 64 + ch * 8,
                    &Qs[row][ch * 8]);
#else
    *(bf16x8*)&Qs[row][ch * 8] =
        *(const bf16x8*)(Q + base + (size_t)(q0 + row) * 64 + ch * 8);
#endif
  }

  f32x8 o[4] = {};
  float m_run = -1e30f, l_run = 0.f;
#if HAVE_ASYNC_LDS
  wait_async0();
#endif
  __syncthreads();

  for (int jb = 0; jb < 32; ++jb) {
    const int kv0 = jb * 64;
    // stage K (row-major, async if available) and V (transposed manually)
#pragma unroll
    for (int i = 0; i < 2; ++i) {
      int c = tid + i * 256;
      int row = c >> 3, ch = c & 7;
#if HAVE_ASYNC_LDS
      async_load_b128(Kk + base + (size_t)(kv0 + row) * 64 + ch * 8,
                      &Ks[row][ch * 8]);
#else
      *(bf16x8*)&Ks[row][ch * 8] =
          *(const bf16x8*)(Kk + base + (size_t)(kv0 + row) * 64 + ch * 8);
#endif
      bf16x8 vv = *(const bf16x8*)(V + base + (size_t)(kv0 + row) * 64 + ch * 8);
#pragma unroll
      for (int j = 0; j < 8; ++j) Vt[ch * 8 + j][row] = vv[j];
    }
#if HAVE_ASYNC_LDS
    wait_async0();
#endif
    __syncthreads();

    // S^T (64kv x 16q) = K_tile(64x64) @ Q^T : K-dim = d = 64, 2 steps of 32
    f32x8 sf[4] = {};
#pragma unroll
    for (int ks = 0; ks < 2; ++ks) {
      const int kk = ks * 32;
      const __bf16* pq = &Qs[wave * 16 + l15][kk + half * 8];
      bf16x16 bq = join8(*(const bf16x8*)pq, *(const bf16x8*)(pq + 16));
#pragma unroll
      for (int f = 0; f < 4; ++f) {
        const __bf16* pk = &Ks[f * 16 + l15][kk + half * 8];
        bf16x16 ak = join8(*(const bf16x8*)pk, *(const bf16x8*)(pk + 16));
        sf[f] = wmma_bf16(ak, bq, sf[f]);
      }
    }

    // online softmax over kv (rows of S^T): per lane + one half-swap shuffle
    float mloc = -1e30f;
#pragma unroll
    for (int f = 0; f < 4; ++f)
#pragma unroll
      for (int r = 0; r < 8; ++r) mloc = fmaxf(mloc, sf[f][r]);
    mloc = fmaxf(mloc, __shfl_xor(mloc, 16, 32));
    const float mnew = fmaxf(m_run, mloc);
    const float corr = __expf(m_run - mnew);
    float lsum = 0.f;
#pragma unroll
    for (int f = 0; f < 4; ++f)
#pragma unroll
      for (int r = 0; r < 8; ++r) {
        float p = __expf(sf[f][r] - mnew);
        sf[f][r] = p;
        lsum += p;
      }
    lsum += __shfl_xor(lsum, 16, 32);
    l_run = l_run * corr + lsum;
    m_run = mnew;
#pragma unroll
    for (int f = 0; f < 4; ++f) o[f] = o[f] * corr;

    // write P (bf16) as [q][kv] into this wave's LDS slab
#pragma unroll
    for (int f = 0; f < 4; ++f)
#pragma unroll
      for (int r = 0; r < 8; r += 2) {
        bf16x2 pp = {(__bf16)sf[f][r], (__bf16)sf[f][r + 1]};
        *(bf16x2*)&Ps[wave][l15][f * 16 + half * 8 + r] = pp;
      }

    // O^T (64d x 16q) += V^T(64x64) @ P^T : K-dim = kv = 64, 2 steps of 32
#pragma unroll
    for (int ks = 0; ks < 2; ++ks) {
      const int kk = ks * 32;
      const __bf16* pp = &Ps[wave][l15][kk + half * 8];
      bf16x16 bp = join8(*(const bf16x8*)pp, *(const bf16x8*)(pp + 16));
#pragma unroll
      for (int f = 0; f < 4; ++f) {
        const __bf16* pv = &Vt[f * 16 + l15][kk + half * 8];
        bf16x16 av = join8(*(const bf16x8*)pv, *(const bf16x8*)(pv + 16));
        o[f] = wmma_bf16(av, bp, o[f]);
      }
    }
    __syncthreads();  // protect Ks/Vt before next block overwrites them
  }

  const float inv = 1.f / l_run;
  const int q = q0 + wave * 16 + l15;
#pragma unroll
  for (int f = 0; f < 4; ++f)
#pragma unroll
    for (int r = 0; r < 8; r += 2) {
      const int d = f * 16 + half * 8 + r;
      bf16x2 oo = {(__bf16)(o[f][r] * inv), (__bf16)(o[f][r + 1] * inv)};
      *(bf16x2*)(O + base + (size_t)q * 64 + d) = oo;
    }
}

// ---------------- host side ----------------
extern "C" void kernel_launch(void* const* d_in, const int* in_sizes, int n_in,
                              void* d_out, int out_size, void* d_ws,
                              size_t ws_size, hipStream_t stream) {
  const float* x = (const float*)d_in[0];
  const float* Wq = (const float*)d_in[1];
  const float* Wk = (const float*)d_in[2];
  const float* Wv = (const float*)d_in[3];
  const float* Wu = (const float*)d_in[4];
  const float* bu = (const float*)d_in[5];
  float* out = (float*)d_out;

  char* ws = (char*)d_ws;
  __bf16* xb  = (__bf16*)(ws + ((size_t)0  << 20));  // 8 MB  [4096,1024]
  __bf16* wqb = (__bf16*)(ws + ((size_t)8  << 20));  // 2 MB
  __bf16* wkb = (__bf16*)(ws + ((size_t)10 << 20));  // 2 MB
  __bf16* wvb = (__bf16*)(ws + ((size_t)12 << 20));  // 2 MB
  __bf16* wub = (__bf16*)(ws + ((size_t)14 << 20));  // 2 MB
  __bf16* Qw  = (__bf16*)(ws + ((size_t)16 << 20));  // 8 MB  [b,h,t,64]
  __bf16* Kw  = (__bf16*)(ws + ((size_t)24 << 20));  // 8 MB
  __bf16* Vw  = (__bf16*)(ws + ((size_t)32 << 20));  // 8 MB
  __bf16* AO  = (__bf16*)(ws + ((size_t)40 << 20));  // 8 MB

  // fp32 -> bf16 conversions
  {
    int n4x = (4096 * 1024) / 4;
    cvt_f32_to_bf16_x4<<<(n4x + 255) / 256, 256, 0, stream>>>(
        (const float4*)x, (bf16x4*)xb, n4x);
    int n4w = (1024 * 1024) / 4;
    cvt_f32_to_bf16_x4<<<(n4w + 255) / 256, 256, 0, stream>>>(
        (const float4*)Wq, (bf16x4*)wqb, n4w);
    cvt_f32_to_bf16_x4<<<(n4w + 255) / 256, 256, 0, stream>>>(
        (const float4*)Wk, (bf16x4*)wkb, n4w);
    cvt_f32_to_bf16_x4<<<(n4w + 255) / 256, 256, 0, stream>>>(
        (const float4*)Wv, (bf16x4*)wvb, n4w);
    cvt_f32_to_bf16_x4<<<(n4w + 255) / 256, 256, 0, stream>>>(
        (const float4*)Wu, (bf16x4*)wub, n4w);
  }

  dim3 gg(1024 / 128, 4096 / 128), bb(256);
  // Q/K/V projections (Q gets the 1/sqrt(64) scale folded into its epilogue)
  gemm_bf16_k<0, 0><<<gg, bb, 0, stream>>>(xb, wqb, Qw, nullptr, nullptr, 0.125f);
  gemm_bf16_k<0, 0><<<gg, bb, 0, stream>>>(xb, wkb, Kw, nullptr, nullptr, 1.0f);
  gemm_bf16_k<0, 0><<<gg, bb, 0, stream>>>(xb, wvb, Vw, nullptr, nullptr, 1.0f);

  // attention: 32 (b*h) x 16 q-blocks of 128 rows
  attn_kernel<<<dim3(32, 2048 / 128), 256, 0, stream>>>(Qw, Kw, Vw, AO);

  // output projection + bias, fp32 result
  gemm_bf16_k<1, 1><<<gg, bb, 0, stream>>>(AO, wub, nullptr, out, bu, 1.0f);
}
